// Attention_20469814133542
// MI455X (gfx1250) — matrix operand order repaired and checked by reading, not compiled, not run on hardware
//
#include <hip/hip_runtime.h>
#include <cstdint>

// ---------- types ----------
typedef _Float16 v16h __attribute__((ext_vector_type(16)));
typedef float    v8f  __attribute__((ext_vector_type(8)));
typedef float    v4f  __attribute__((ext_vector_type(4)));
typedef int      v4i  __attribute__((ext_vector_type(4)));

union Frag { v16h h; v4f q[2]; };

__device__ __forceinline__ v8f zero8() {
  v8f z = {0.f, 0.f, 0.f, 0.f, 0.f, 0.f, 0.f, 0.f};
  return z;
}

__device__ __forceinline__ v8f wmma_f16(const Frag& a, const Frag& b, v8f c) {
  // D = A(16x32 f16) * B(32x16 f16) + C(16x16 f32)
  return __builtin_amdgcn_wmma_f32_16x16x32_f16(
      /*neg_a=*/false, a.h, /*neg_b=*/false, b.h,
      /*c_mod=*/(short)0, c, /*reuse_a=*/false, /*reuse_b=*/false);
}

// ---------- async global->LDS copy (CDNA5 path with fallback) ----------
#if defined(__AMDGCN__) && __has_builtin(__builtin_amdgcn_global_load_async_to_lds_b128)
#define USE_ASYNC 1
#else
#define USE_ASYNC 0
#endif

#if USE_ASYNC
typedef __attribute__((address_space(1))) v4i g128_t;  // global 16B vector
typedef __attribute__((address_space(3))) v4i l128_t;  // LDS 16B vector
#endif

__device__ __forceinline__ void cp16(const _Float16* g, _Float16* l) {
#if USE_ASYNC
  __builtin_amdgcn_global_load_async_to_lds_b128((g128_t*)g, (l128_t*)l, 0, 0);
#else
  *(v4f*)l = *(const v4f*)g;
#endif
}

template <int N>
__device__ __forceinline__ void waitAsync() {
#if USE_ASYNC
#if __has_builtin(__builtin_amdgcn_s_wait_asynccnt)
  __builtin_amdgcn_s_wait_asynccnt(N);
#else
  if (N == 0) asm volatile("s_wait_asynccnt 0" ::: "memory");
  else        asm volatile("s_wait_asynccnt 4" ::: "memory");
#endif
#endif
}

// ---------- DPP16 butterfly reductions over a 16-lane half (no LDS traffic) ----------
// xor1 = quad_perm(1,0,3,2)=0xB1, xor2 = quad_perm(2,3,0,1)=0x4E,
// xor7 = row_half_mirror=0x141, xor15 = row_mirror=0x140.
template <int CTRL>
__device__ __forceinline__ float dppf(float v) {
  return __int_as_float(__builtin_amdgcn_update_dpp(
      0, __float_as_int(v), CTRL, 0xF, 0xF, true));
}
__device__ __forceinline__ float redMax16(float v) {
  v = fmaxf(v, dppf<0xB1>(v));
  v = fmaxf(v, dppf<0x4E>(v));
  v = fmaxf(v, dppf<0x141>(v));
  v = fmaxf(v, dppf<0x140>(v));
  return v;
}
__device__ __forceinline__ float redSum16(float v) {
  v += dppf<0xB1>(v);
  v += dppf<0x4E>(v);
  v += dppf<0x141>(v);
  v += dppf<0x140>(v);
  return v;
}

// ---------- constants ----------
constexpr int EMB = 1024;     // E = K = N of projections
constexpr int SEQ = 2048;
constexpr int BATCH = 2;
constexpr int HEADS = 16;
constexpr int HD = 64;
constexpr int MROWS = BATCH * SEQ;      // 4096
constexpr int LDSTR = 40;               // 32 halves + 8 pad (80B, 16B-aligned)

// ---------- f32 -> f16 convert ----------
__global__ void f32_to_f16(const float* __restrict__ in, _Float16* __restrict__ out, int n) {
  int i = blockIdx.x * 256 + threadIdx.x;
  if (i < n) out[i] = (_Float16)in[i];
}

// ---------- WMMA GEMM: Y = A[M,K] * W[N,K]^T + bias, tiled 128x128 ----------
// MODE 0: f32 row-major [M,N] out.  MODE 1: f16 [B,H,S,D] out (QKV), *scale.
// MODE 2: f16 [B,H,D,S] out (V transposed).
__device__ __forceinline__ void issueTile(const _Float16* A, const _Float16* W,
                                          int mbase, int nbase, int k0,
                                          _Float16* la, _Float16* lw, int tid) {
#pragma unroll
  for (int i = 0; i < 2; ++i) {
    int c = tid + i * 256;          // 512 chunks of 16B per tile
    int row = c >> 2, sub = c & 3;  // sub: which 8-half chunk in the 32-wide K slab
    cp16(A + (size_t)(mbase + row) * EMB + k0 + sub * 8, la + row * LDSTR + sub * 8);
    cp16(W + (size_t)(nbase + row) * EMB + k0 + sub * 8, lw + row * LDSTR + sub * 8);
  }
}

template <int MODE>
__global__ __launch_bounds__(256) void gemm_kernel(const _Float16* __restrict__ A,
                                                   const _Float16* __restrict__ W,
                                                   const float* __restrict__ bias,
                                                   void* __restrict__ outp,
                                                   float scale) {
  __shared__ _Float16 lA[2][128 * LDSTR];
  __shared__ _Float16 lW[2][128 * LDSTR];
  const int tid = threadIdx.x;
  const int wv = tid >> 5, lane = tid & 31;
  const int wM = wv >> 2, wN = wv & 3;      // 2x4 wave grid -> 64x32 per wave
  const int l16 = lane & 15, lh = lane >> 4;
  const int mbase = blockIdx.y * 128, nbase = blockIdx.x * 128;
  const int NK = EMB / 32;

  v8f acc[4][2];
#pragma unroll
  for (int mt = 0; mt < 4; ++mt)
#pragma unroll
    for (int nt = 0; nt < 2; ++nt) acc[mt][nt] = zero8();

  issueTile(A, W, mbase, nbase, 0, lA[0], lW[0], tid);

  for (int kk = 0; kk < NK; ++kk) {
    const int buf = kk & 1;
    if (kk + 1 < NK) {
      issueTile(A, W, mbase, nbase, (kk + 1) * 32, lA[buf ^ 1], lW[buf ^ 1], tid);
      waitAsync<4>();   // oldest 4 (this buffer's) async ops done
    } else {
      waitAsync<0>();
    }
    __syncthreads();

    Frag af[4];
#pragma unroll
    for (int mt = 0; mt < 4; ++mt) {
      const _Float16* p = &lA[buf][(wM * 64 + mt * 16 + l16) * LDSTR + lh * 8];
      af[mt].q[0] = *(const v4f*)p;
      af[mt].q[1] = *(const v4f*)(p + 16);
    }
    Frag bf[2];
#pragma unroll
    for (int nt = 0; nt < 2; ++nt) {
      const _Float16* p = &lW[buf][(wN * 32 + nt * 16 + l16) * LDSTR + lh * 8];
      bf[nt].q[0] = *(const v4f*)p;
      bf[nt].q[1] = *(const v4f*)(p + 16);
    }
#pragma unroll
    for (int mt = 0; mt < 4; ++mt)
#pragma unroll
      for (int nt = 0; nt < 2; ++nt)
        acc[mt][nt] = wmma_f16(af[mt], bf[nt], acc[mt][nt]);
    __syncthreads();
  }

  // epilogue: D layout = VGPR r holds rows r (lanes 0-15) / r+8 (lanes 16-31), col = lane%16
#pragma unroll
  for (int nt = 0; nt < 2; ++nt) {
    const int col = nbase + wN * 32 + nt * 16 + l16;
    const float bv = bias[col];
#pragma unroll
    for (int mt = 0; mt < 4; ++mt) {
#pragma unroll
      for (int r = 0; r < 8; ++r) {
        const int row = mbase + wM * 64 + mt * 16 + r + 8 * lh;
        const float v = (acc[mt][nt][r] + bv) * scale;
        if (MODE == 0) {
          ((float*)outp)[(size_t)row * EMB + col] = v;
        } else {
          const int b = row >> 11, ss = row & (SEQ - 1);
          const int h = col >> 6, d = col & (HD - 1);
          if (MODE == 1)
            ((_Float16*)outp)[(((size_t)b * HEADS + h) * SEQ + ss) * HD + d] = (_Float16)v;
          else
            ((_Float16*)outp)[(((size_t)b * HEADS + h) * HD + d) * SEQ + ss] = (_Float16)v;
        }
      }
    }
  }
}

// ---------- flash attention: softmax(Q K^T) V, scale already folded into Q ----------
__global__ __launch_bounds__(256) void attn_kernel(const _Float16* __restrict__ Q,
                                                   const _Float16* __restrict__ Km,
                                                   const _Float16* __restrict__ Vt,
                                                   _Float16* __restrict__ Out) {
  __shared__ _Float16 plds[8][16 * 72];  // per-wave 16x64 P tile, stride 72 halves (144B)
  const int tid = threadIdx.x, wv = tid >> 5, lane = tid & 31;
  const int l16 = lane & 15, lh = lane >> 4;
  const int bh = blockIdx.x, qb = blockIdx.y;
  const int qrow0 = qb * 128 + wv * 16;
  const _Float16* Qb = Q + (size_t)bh * SEQ * HD;
  const _Float16* Kb = Km + (size_t)bh * SEQ * HD;
  const _Float16* Vb = Vt + (size_t)bh * HD * SEQ;
  _Float16* pw = &plds[wv][0];

  // Q fragments for this wave's 16 rows: d 0..31 and 32..63
  Frag qf0, qf1;
  {
    const _Float16* p = Qb + (size_t)(qrow0 + l16) * HD + lh * 8;
    qf0.q[0] = *(const v4f*)p;        qf0.q[1] = *(const v4f*)(p + 16);
    qf1.q[0] = *(const v4f*)(p + 32); qf1.q[1] = *(const v4f*)(p + 48);
  }

  float m[8], l[8];
  v8f o[4];
#pragma unroll
  for (int r = 0; r < 8; ++r) { m[r] = -1e30f; l[r] = 0.f; }
#pragma unroll
  for (int j = 0; j < 4; ++j) o[j] = zero8();

  for (int kt = 0; kt < SEQ / 64; ++kt) {  // 64-key tiles
    v8f s[4];
#pragma unroll
    for (int t = 0; t < 4; ++t) {
      const int kr = kt * 64 + t * 16 + l16;  // key index for this lane's B column
      const _Float16* p = Kb + (size_t)kr * HD + lh * 8;
      Frag k0f, k1f;
      k0f.q[0] = *(const v4f*)p;        k0f.q[1] = *(const v4f*)(p + 16);
      k1f.q[0] = *(const v4f*)(p + 32); k1f.q[1] = *(const v4f*)(p + 48);
      s[t] = wmma_f16(qf0, k0f, zero8());
      s[t] = wmma_f16(qf1, k1f, s[t]);
    }

    float nm[8], corr[8];
#pragma unroll
    for (int r = 0; r < 8; ++r) {
      float tm = fmaxf(fmaxf(s[0][r], s[1][r]), fmaxf(s[2][r], s[3][r]));
      tm = redMax16(tm);
      nm[r] = fmaxf(m[r], tm);
      corr[r] = __expf(m[r] - nm[r]);
      m[r] = nm[r];
    }
#pragma unroll
    for (int t = 0; t < 4; ++t)
#pragma unroll
      for (int r = 0; r < 8; ++r) s[t][r] = __expf(s[t][r] - nm[r]);
#pragma unroll
    for (int r = 0; r < 8; ++r) {
      float rs = redSum16(s[0][r] + s[1][r] + s[2][r] + s[3][r]);
      l[r] = l[r] * corr[r] + rs;
    }
#pragma unroll
    for (int j = 0; j < 4; ++j)
#pragma unroll
      for (int r = 0; r < 8; ++r) o[j][r] *= corr[r];

    // P (C-layout) -> LDS row-major, then reload as A-layout fragments
    asm volatile("" ::: "memory");
#pragma unroll
    for (int t = 0; t < 4; ++t)
#pragma unroll
      for (int r = 0; r < 8; ++r)
        pw[(r + 8 * lh) * 72 + (t * 16 + l16)] = (_Float16)s[t][r];
    asm volatile("s_wait_dscnt 0" ::: "memory");

#pragma unroll
    for (int kc = 0; kc < 2; ++kc) {  // key chunks of 32
      Frag pf;
      const _Float16* pp = pw + l16 * 72 + kc * 32 + lh * 8;
      pf.q[0] = *(const v4f*)pp;
      pf.q[1] = *(const v4f*)(pp + 16);
#pragma unroll
      for (int j = 0; j < 4; ++j) {  // d tiles of 16
        Frag vf;
        const _Float16* vp = Vb + (size_t)(j * 16 + l16) * SEQ + kt * 64 + kc * 32 + lh * 8;
        vf.q[0] = *(const v4f*)vp;
        vf.q[1] = *(const v4f*)(vp + 16);
        o[j] = wmma_f16(pf, vf, o[j]);
      }
    }
    asm volatile("s_wait_dscnt 0" ::: "memory");
  }

  const int b = bh >> 4, h = bh & (HEADS - 1);
#pragma unroll
  for (int r = 0; r < 8; ++r) {
    const float inv = 1.f / l[r];
    const int srow = qrow0 + r + 8 * lh;
#pragma unroll
    for (int j = 0; j < 4; ++j) {
      const int d = j * 16 + l16;
      Out[(((size_t)b * SEQ + srow) * HEADS + h) * HD + d] = (_Float16)(o[j][r] * inv);
    }
  }
}

// ---------- host launcher ----------
extern "C" void kernel_launch(void* const* d_in, const int* in_sizes, int n_in,
                              void* d_out, int out_size, void* d_ws, size_t ws_size,
                              hipStream_t stream) {
  (void)in_sizes; (void)n_in; (void)out_size; (void)ws_size;
  const float* x  = (const float*)d_in[0];
  const float* Wq = (const float*)d_in[1];
  const float* bq = (const float*)d_in[2];
  const float* Wk = (const float*)d_in[3];
  const float* bk = (const float*)d_in[4];
  const float* Wv = (const float*)d_in[5];
  const float* bv = (const float*)d_in[6];
  const float* Wo = (const float*)d_in[7];
  const float* bo = (const float*)d_in[8];

  char* ws = (char*)d_ws;
  const size_t xBytes = (size_t)MROWS * EMB * 2;     // 8 MB
  const size_t wBytes = (size_t)EMB * EMB * 2;       // 2 MB each
  const size_t hBytes = (size_t)BATCH * HEADS * SEQ * HD * 2;  // 8 MB each
  _Float16* xh  = (_Float16*)ws;            ws += xBytes;
  _Float16* wqh = (_Float16*)ws;            ws += wBytes;
  _Float16* wkh = (_Float16*)ws;            ws += wBytes;
  _Float16* wvh = (_Float16*)ws;            ws += wBytes;
  _Float16* woh = (_Float16*)ws;            ws += wBytes;
  _Float16* qh  = (_Float16*)ws;            ws += hBytes;
  _Float16* kh  = (_Float16*)ws;            ws += hBytes;
  _Float16* vth = (_Float16*)ws;            ws += hBytes;
  _Float16* ah  = (_Float16*)ws;            ws += hBytes;

  const int nX = MROWS * EMB, nW = EMB * EMB;
  f32_to_f16<<<(nX + 255) / 256, 256, 0, stream>>>(x,  xh,  nX);
  f32_to_f16<<<(nW + 255) / 256, 256, 0, stream>>>(Wq, wqh, nW);
  f32_to_f16<<<(nW + 255) / 256, 256, 0, stream>>>(Wk, wkh, nW);
  f32_to_f16<<<(nW + 255) / 256, 256, 0, stream>>>(Wv, wvh, nW);
  f32_to_f16<<<(nW + 255) / 256, 256, 0, stream>>>(Wo, woh, nW);

  dim3 ggrid(EMB / 128, MROWS / 128);  // (8, 32)
  const float scaleQ = 0.125f;         // 1/sqrt(64), folded into Q
  gemm_kernel<1><<<ggrid, 256, 0, stream>>>(xh, wqh, bq, qh,  scaleQ);
  gemm_kernel<1><<<ggrid, 256, 0, stream>>>(xh, wkh, bk, kh,  1.0f);
  gemm_kernel<2><<<ggrid, 256, 0, stream>>>(xh, wvh, bv, vth, 1.0f);

  attn_kernel<<<dim3(BATCH * HEADS, SEQ / 128), 256, 0, stream>>>(qh, kh, vth, ah);

  gemm_kernel<0><<<ggrid, 256, 0, stream>>>(ah, woh, bo, d_out, 1.0f);
}
